// DiscreteAttention_87007447483023
// MI455X (gfx1250) — matrix operand order repaired and checked by reading, not compiled, not run on hardware
//
#include <hip/hip_runtime.h>
#include <hip/hip_bf16.h>
#include <math.h>

// ---------------------------------------------------------------------------
// 1.5-entmax attention for MI455X (gfx1250), wave32 + v_wmma_f32_16x16x32_bf16
// Pipeline:
//   0a) Q,K  -> bf16 copies (ws)            0b) V -> bf16 TRANSPOSED copy (ws)
//   1) scores = mask(scale * Qbf.Kbf^T) -> written into p_attn region of d_out
//   2) exact sort-based 1.5-entmax in place on each row; bf16 copy of p to ws
//   3) o = Pbf @ V  using s-major Vt  -> o region of d_out
// GEMM waves own a 32x64 strip: 2 A-frags x 4 B-frags -> 8 WMMA burst per
// 32-wide K step (B-fragments reused 2x; single loadcnt wait per burst).
// ---------------------------------------------------------------------------

typedef __attribute__((ext_vector_type(16))) __bf16 v16bf;
typedef __attribute__((ext_vector_type(8)))  __bf16 v8bf;
typedef __attribute__((ext_vector_type(4)))  __bf16 v4bf;
typedef __attribute__((ext_vector_type(8)))  float  v8f;

static constexpr int   Bn   = 8;
static constexpr int   Tn   = 2048;
static constexpr int   Sn   = 2048;
static constexpr int   Hn   = 512;
static constexpr float NEGV = -1e4f;

// ===========================================================================
// Prologue 0a: elementwise fp32 -> bf16 (Q, K)
// ===========================================================================
__global__ __launch_bounds__(256) void cvt_bf16_kernel(
    const float* __restrict__ src, __bf16* __restrict__ dst, int n)
{
    const int i = (blockIdx.x * 256 + threadIdx.x) * 4;
    if (i < n) {
        const float4 v = *(const float4*)(src + i);
        v4bf o;
        o[0] = (__bf16)v.x; o[1] = (__bf16)v.y;
        o[2] = (__bf16)v.z; o[3] = (__bf16)v.w;
        *(v4bf*)(dst + i) = o;
    }
}

// ===========================================================================
// Prologue 0b: V (B,S,H) fp32 -> Vt (B,H,S) bf16, 32x32 LDS tile transpose
// ===========================================================================
__global__ __launch_bounds__(256) void vtrans_bf16_kernel(
    const float* __restrict__ V, __bf16* __restrict__ Vt)
{
    __shared__ float tile[32][33];
    const int b  = blockIdx.z;
    const int h0 = blockIdx.x * 32;
    const int s0 = blockIdx.y * 32;
    const int tx = threadIdx.x & 31;
    const int ty = threadIdx.x >> 5;            // 8 rows per pass

    #pragma unroll
    for (int r = 0; r < 32; r += 8)
        tile[ty + r][tx] = V[((size_t)b * Sn + s0 + ty + r) * Hn + h0 + tx];
    __syncthreads();
    #pragma unroll
    for (int r = 0; r < 32; r += 8)
        Vt[((size_t)b * Hn + h0 + ty + r) * Sn + s0 + tx] =
            (__bf16)tile[tx][ty + r];
}

// ===========================================================================
// Kernel 1: scores[b,t,s] = scale * sum_h Qbf[b,t,h]*Kbf[b,s,h], masked.
// Block 256 threads (8 waves), tile 128(T) x 128(S); wave strip 32 x 64.
// ===========================================================================
__global__ __launch_bounds__(256) void scores_wmma_kernel(
    const __bf16* __restrict__ Q, const __bf16* __restrict__ K,
    const int* __restrict__ mask, float* __restrict__ S)
{
    const float scale = 0.044194173824159216f;  // 1/sqrt(512)
    const int b  = blockIdx.z;
    const int t0 = blockIdx.y * 128;
    const int s0 = blockIdx.x * 128;

    const int wave = threadIdx.x >> 5;
    const int lane = threadIdx.x & 31;
    const int wr   = wave & 3;          // 0..3 -> 32-row strips
    const int wc   = wave >> 2;         // 0..1 -> 64-col strips
    const int m0   = t0 + wr * 32;
    const int n0   = s0 + wc * 64;

    const int lrow = lane & 15;
    const int khA  = (lane >> 4) * 16;  // A layout: K groups of 16
    const int khB  = (lane >> 4) * 16;  // B layout: K groups of 16

    const __bf16* Qb = Q + ((size_t)b * Tn + m0) * Hn;
    const __bf16* Kb = K + ((size_t)b * Sn + n0) * Hn;

    v8f acc[2][4] = {};

    #pragma unroll 2
    for (int k0 = 0; k0 < Hn; k0 += 32) {
        v16bf afrag[2];
        #pragma unroll
        for (int a = 0; a < 2; ++a) {
            const __bf16* qp = Qb + (size_t)(a * 16 + lrow) * Hn + k0 + khA;
            __builtin_prefetch(qp + 64, 0, 1);
            const v8bf lo = *(const v8bf*)qp;
            const v8bf hi = *(const v8bf*)(qp + 8);
            #pragma unroll
            for (int i = 0; i < 8; ++i) {
                afrag[a][i] = lo[i]; afrag[a][8 + i] = hi[i];
            }
        }
        v16bf bfrag[4];
        #pragma unroll
        for (int nt = 0; nt < 4; ++nt) {
            const __bf16* kp = Kb + (size_t)(nt * 16 + lrow) * Hn + k0 + khB;
            __builtin_prefetch(kp + 64, 0, 1);
            bfrag[nt] = *(const v16bf*)kp;
        }
        #pragma unroll
        for (int a = 0; a < 2; ++a)
            #pragma unroll
            for (int nt = 0; nt < 4; ++nt)
                acc[a][nt] = __builtin_amdgcn_wmma_f32_16x16x32_bf16(
                    false, afrag[a], false, bfrag[nt], (short)0,
                    acc[a][nt], false, false);
    }

    // store with scale + mask (C/D layout: N = lane%16, M = r / r+8)
    float* Sp = S + (size_t)b * Tn * Sn;
    const int hhalf = (lane >> 4) ? 8 : 0;
    #pragma unroll
    for (int nt = 0; nt < 4; ++nt) {
        const int s  = n0 + nt * 16 + lrow;
        const int mk = mask[(size_t)b * Sn + s];
        #pragma unroll
        for (int a = 0; a < 2; ++a) {
            const int mbase = m0 + a * 16 + hhalf;
            #pragma unroll
            for (int r = 0; r < 8; ++r) {
                const float v = mk ? acc[a][nt][r] * scale : NEGV;
                Sp[(size_t)(mbase + r) * Sn + s] = v;
            }
        }
    }
}

// ===========================================================================
// Kernel 2: exact sort-based 1.5-entmax, in place on each row of P.
// One 256-thread block per row; also emits bf16 copy of p for kernel 3.
// ===========================================================================
__global__ __launch_bounds__(256) void entmax15_kernel(
    float* __restrict__ P, __bf16* __restrict__ Pbf)
{
    __shared__ float xo[Sn];       // original (unsorted) shifted x
    __shared__ float xs[Sn];       // sorted (desc) shifted x
    __shared__ float cs[Sn];       // cumsum(xs)
    __shared__ float cq[Sn];       // cumsum(xs^2), later tau
    __shared__ float ps[256], pq[256], red[256];
    __shared__ int   scount;
    __shared__ float stau;

    const size_t row = blockIdx.x;                  // b*T + t
    float*       z   = P + row * (size_t)Sn;
    const int    tid = threadIdx.x;

    if (tid == 0) scount = 0;

    // x = z/2 ; row max
    float lmax = -INFINITY;
    for (int i = tid; i < Sn; i += 256) {
        const float v = z[i] * 0.5f;
        xo[i] = v;
        lmax  = fmaxf(lmax, v);
    }
    red[tid] = lmax;
    __syncthreads();
    #pragma unroll
    for (int off = 128; off > 0; off >>= 1) {
        if (tid < off) red[tid] = fmaxf(red[tid], red[tid + off]);
        __syncthreads();
    }
    const float mx = red[0];
    for (int i = tid; i < Sn; i += 256) {
        const float v = xo[i] - mx;
        xo[i] = v;
        xs[i] = v;
    }
    __syncthreads();

    // bitonic sort, descending
    for (int k = 2; k <= Sn; k <<= 1) {
        for (int j = k >> 1; j > 0; j >>= 1) {
            for (int i = tid; i < Sn; i += 256) {
                const int ixj = i ^ j;
                if (ixj > i) {
                    const float a = xs[i], c = xs[ixj];
                    const bool desc = ((i & k) == 0);
                    if ((a < c) == desc) { xs[i] = c; xs[ixj] = a; }
                }
            }
            __syncthreads();
        }
    }

    // 3-phase inclusive scan of xs and xs^2 (8 elements per thread)
    const int base = tid * (Sn / 256);
    float s1 = 0.f, s2 = 0.f;
    #pragma unroll
    for (int u = 0; u < Sn / 256; ++u) {
        const float v = xs[base + u];
        s1 += v; s2 += v * v;
    }
    ps[tid] = s1; pq[tid] = s2;
    __syncthreads();
    if (tid == 0) {
        float a1 = 0.f, a2 = 0.f;
        for (int t = 0; t < 256; ++t) {
            const float b1 = ps[t], b2 = pq[t];
            ps[t] = a1; pq[t] = a2;
            a1 += b1; a2 += b2;
        }
    }
    __syncthreads();
    float r1 = ps[tid], r2 = pq[tid];
    #pragma unroll
    for (int u = 0; u < Sn / 256; ++u) {
        const float v = xs[base + u];
        r1 += v; r2 += v * v;
        cs[base + u] = r1;
        cq[base + u] = r2;
    }

    // tau_k = mean_k - sqrt(clip((1-ss_k)/k)); support count (xs_k >= tau_k)
    int lcount = 0;
    #pragma unroll
    for (int u = 0; u < Sn / 256; ++u) {
        const int   i      = base + u;
        const float kf     = (float)(i + 1);
        const float mean   = cs[i] / kf;
        const float meansq = cq[i] / kf;
        const float ss     = kf * (meansq - mean * mean);
        const float delta  = (1.0f - ss) / kf;
        const float tau    = mean - sqrtf(fmaxf(delta, 0.0f));
        cq[i] = tau;                       // own slot only: safe overwrite
        if (xs[i] >= tau) ++lcount;
    }
    __syncthreads();
    atomicAdd(&scount, lcount);
    __syncthreads();
    if (tid == 0) stau = cq[scount - 1];   // tau_star at k_star = count-1
    __syncthreads();
    const float tstar = stau;

    // p = clip(x - tau_star, 0)^2 ; write f32 (output) + bf16 (for kernel 3)
    __bf16* zb = Pbf + row * (size_t)Sn;
    for (int i = tid; i < Sn; i += 256) {
        const float d = fmaxf(xo[i] - tstar, 0.0f);
        const float p = d * d;
        z[i]  = p;
        zb[i] = (__bf16)p;
    }
}

// ===========================================================================
// Kernel 3: o[b,t,h] = sum_s Pbf[b,t,s] * Vt[b,h,s]  (both operands s-major)
// Same tiling as kernel 1: 128(T) x 128(H) per block, wave strip 32 x 64.
// ===========================================================================
__global__ __launch_bounds__(256) void out_wmma_kernel(
    const __bf16* __restrict__ Pbf, const __bf16* __restrict__ Vt,
    float* __restrict__ O)
{
    const int b  = blockIdx.z;
    const int t0 = blockIdx.y * 128;
    const int h0 = blockIdx.x * 128;

    const int wave = threadIdx.x >> 5;
    const int lane = threadIdx.x & 31;
    const int wr   = wave & 3;
    const int wc   = wave >> 2;
    const int m0   = t0 + wr * 32;
    const int n0   = h0 + wc * 64;

    const int lrow = lane & 15;
    const int khA  = (lane >> 4) * 16;
    const int khB  = (lane >> 4) * 16;

    const __bf16* Pb = Pbf + ((size_t)b * Tn + m0) * Sn;
    const __bf16* Vb = Vt  + ((size_t)b * Hn + n0) * Sn;

    v8f acc[2][4] = {};

    #pragma unroll 2
    for (int k0 = 0; k0 < Sn; k0 += 32) {
        v16bf afrag[2];
        #pragma unroll
        for (int a = 0; a < 2; ++a) {
            const __bf16* pp = Pb + (size_t)(a * 16 + lrow) * Sn + k0 + khA;
            __builtin_prefetch(pp + 64, 0, 1);
            const v8bf lo = *(const v8bf*)pp;
            const v8bf hi = *(const v8bf*)(pp + 8);
            #pragma unroll
            for (int i = 0; i < 8; ++i) {
                afrag[a][i] = lo[i]; afrag[a][8 + i] = hi[i];
            }
        }
        v16bf bfrag[4];
        #pragma unroll
        for (int nt = 0; nt < 4; ++nt) {
            const __bf16* vp = Vb + (size_t)(nt * 16 + lrow) * Sn + k0 + khB;
            __builtin_prefetch(vp + 64, 0, 1);
            bfrag[nt] = *(const v16bf*)vp;
        }
        #pragma unroll
        for (int a = 0; a < 2; ++a)
            #pragma unroll
            for (int nt = 0; nt < 4; ++nt)
                acc[a][nt] = __builtin_amdgcn_wmma_f32_16x16x32_bf16(
                    false, afrag[a], false, bfrag[nt], (short)0,
                    acc[a][nt], false, false);
    }

    const int hhalf = (lane >> 4) ? 8 : 0;
    #pragma unroll
    for (int nt = 0; nt < 4; ++nt) {
        const int h = n0 + nt * 16 + lrow;
        #pragma unroll
        for (int a = 0; a < 2; ++a) {
            const int mbase = m0 + a * 16 + hhalf;
            #pragma unroll
            for (int r = 0; r < 8; ++r)
                O[((size_t)b * Tn + mbase + r) * Hn + h] = acc[a][nt][r];
        }
    }
}

// ===========================================================================
extern "C" void kernel_launch(void* const* d_in, const int* in_sizes, int n_in,
                              void* d_out, int out_size, void* d_ws, size_t ws_size,
                              hipStream_t stream) {
    const float* Q    = (const float*)d_in[0];   // (B,T,H)
    const float* K    = (const float*)d_in[1];   // (B,S,H)
    const float* V    = (const float*)d_in[2];   // (B,S,H)
    const int*   mask = (const int*)  d_in[3];   // (B,S)

    float* o_out = (float*)d_out;                        // (B,T,H)
    float* p_out = o_out + (size_t)Bn * Tn * Hn;         // (B,T,S)

    // workspace layout (bf16): Qbf | Kbf | Vt | Pbf
    __bf16* Qbf = (__bf16*)d_ws;
    __bf16* Kbf = Qbf + (size_t)Bn * Tn * Hn;
    __bf16* Vtb = Kbf + (size_t)Bn * Sn * Hn;
    __bf16* Pbf = Vtb + (size_t)Bn * Hn * Sn;

    const int nQ = Bn * Tn * Hn;   // 8,388,608
    const int nK = Bn * Sn * Hn;

    // 0a) Q,K -> bf16
    cvt_bf16_kernel<<<dim3(nQ / 4 / 256), 256, 0, stream>>>(Q, Qbf, nQ);
    cvt_bf16_kernel<<<dim3(nK / 4 / 256), 256, 0, stream>>>(K, Kbf, nK);
    // 0b) V -> bf16 transposed (B,H,S)
    vtrans_bf16_kernel<<<dim3(Hn / 32, Sn / 32, Bn), 256, 0, stream>>>(V, Vtb);

    // 1) masked scaled scores -> p region (in-place buffer for entmax)
    scores_wmma_kernel<<<dim3(Sn / 128, Tn / 128, Bn), 256, 0, stream>>>(
        Qbf, Kbf, mask, p_out);

    // 2) exact 1.5-entmax per row, in place; bf16 copy for kernel 3
    entmax15_kernel<<<dim3(Bn * Tn), 256, 0, stream>>>(p_out, Pbf);

    // 3) o = P @ V via bf16 WMMA on s-major operands
    out_wmma_kernel<<<dim3(Hn / 128, Tn / 128, Bn), 256, 0, stream>>>(
        Pbf, Vtb, o_out);
}
